// Reason_i2t_89507118448647
// MI455X (gfx1250) — matrix operand order repaired and checked by reading, not compiled, not run on hardware
//
#include <hip/hip_runtime.h>

typedef __bf16 bf16;
typedef bf16  v16bf __attribute__((ext_vector_type(16)));
typedef float v8f   __attribute__((ext_vector_type(8)));

#define WMMA_BF16(A,B,C) __builtin_amdgcn_wmma_f32_16x16x32_bf16(false,(A),false,(B),(short)0,(C),false,false)

// A-fragment K order for 16-bit 16x32 A tiles (contiguous, matches B loads):
// lanes 0-15: elements 0..15 -> K=0..15
// lanes16-31: elements 0..15 -> K=16..31
__device__ __forceinline__ int kApat(int g, int e){
  return g * 16 + e;
}

// load A fragment (M=16,K=32, bf16) from an LDS row-major matrix
__device__ __forceinline__ v16bf ldsA(const bf16* p, int rs, int m0, int k0, int lane){
  int g = lane >> 4;
  const bf16* row = p + (size_t)(m0 + (lane & 15)) * rs + k0;
  v16bf a;
  #pragma unroll
  for (int e = 0; e < 16; ++e) a[e] = row[kApat(g, e)];
  return a;
}

// ---------------------------------------------------------------------------
// prep kernel: fold weights into bf16, transposed for contiguous B-fragment
// loads. ws layout (bytes):
//   0     : W1  bf16 [n][k]   (64x64)  B for f1  (B[k=nb][col=n] = W1[n][nb])
//   8192  : W2  bf16 [n][k]   (64x64)
//   16384 : WcT bf16 [m][n]   (64x64)  WcT[m][n] = sum_k conv_W[k][n][m]/8
//   24576 : WmT bf16 [h][n]   (256x64) WmT[h][n] = mean_p Wm[n][h*2+p]
//   57344 : wn  f32  [256]    g*V/||V||
//   58368 : bme f32  [256]    mean_p bm[h*2+p]
// ---------------------------------------------------------------------------
__global__ void i2t_prep(const float* __restrict__ W1, const float* __restrict__ W2,
                         const float* __restrict__ cw, const float* __restrict__ Wm,
                         const float* __restrict__ bm, const float* __restrict__ V,
                         const float* __restrict__ g, char* __restrict__ ws){
  bf16*  w1  = (bf16*)(ws + 0);
  bf16*  w2  = (bf16*)(ws + 8192);
  bf16*  wc  = (bf16*)(ws + 16384);
  bf16*  wm  = (bf16*)(ws + 24576);
  float* wn  = (float*)(ws + 57344);
  float* bme = (float*)(ws + 58368);
  const int t = threadIdx.x;

  for (int idx = t; idx < 4096; idx += 256){
    w1[idx] = (bf16)W1[idx];
    w2[idx] = (bf16)W2[idx];
  }
  for (int idx = t; idx < 4096; idx += 256){
    int m = idx >> 6, n = idx & 63;
    float s = 0.f;
    #pragma unroll
    for (int k = 0; k < 8; ++k) s += cw[k * 4096 + n * 64 + m];
    wc[m * 64 + n] = (bf16)(s * 0.125f);
  }
  for (int idx = t; idx < 16384; idx += 256){
    int h = idx >> 6, n = idx & 63;
    wm[h * 64 + n] = (bf16)(0.5f * (Wm[n * 512 + h * 2] + Wm[n * 512 + h * 2 + 1]));
  }
  __shared__ float red[256];
  red[t] = V[t] * V[t];
  __syncthreads();
  for (int o = 128; o > 0; o >>= 1){
    if (t < o) red[t] += red[t + o];
    __syncthreads();
  }
  float inv = rsqrtf(red[0]);
  wn[t]  = g[0] * V[t] * inv;
  bme[t] = 0.5f * (bm[2 * t] + bm[2 * t + 1]);
}

// ---------------------------------------------------------------------------
// main kernel: one workgroup (256 threads = 8 wave32) per (caption, image)
// ---------------------------------------------------------------------------
#define CAP_RS 1040

__global__ __launch_bounds__(256, 1)
void i2t_main(const float* __restrict__ img, const float* __restrict__ cap,
              const float* __restrict__ delt, const float* __restrict__ b1,
              const float* __restrict__ b2, const float* __restrict__ convb,
              const float* __restrict__ bo, const char* __restrict__ ws,
              float* __restrict__ out){
  __shared__ bf16  cap_sm[32 * CAP_RS];   // caption words, bf16
  __shared__ float a_sm[32 * 48];         // logits a[w][s] (K-split accumulator)
  __shared__ float invn[32];              // 1/(||a[w,:]||+eps)
  __shared__ bf16  attn_sm[48 * 32];      // softmax attn[s][w]
  __shared__ float qn_sm[48 * 64];        // |q|^2 per (s, block)
  __shared__ float w12_sm[48 * 64];       // dot(q, ctx) per (s, block)
  __shared__ float ncb_sm[48 * 64];       // |ctx|^2 per (s, block)
  __shared__ bf16  sim_sm[48 * 64];       // blockwise cosine * delt
  __shared__ bf16  f1_sm[48 * 64];
  __shared__ bf16  f2_sm[48 * 64];
  __shared__ float gram_sm[48 * 52];
  __shared__ bf16  gw_sm[48 * 64];        // softmax adjacency, K-padded
  __shared__ bf16  msg_sm[48 * 64];
  __shared__ bf16  conv_sm[48 * 64];
  __shared__ float ysum[48];
  __shared__ float ytan[48];

  const int ci   = blockIdx.x;   // caption
  const int ii   = blockIdx.y;   // image
  const int tid  = threadIdx.x;
  const int lane = tid & 31;
  const int wv   = tid >> 5;
  const int g    = lane >> 4;
  const int n    = lane & 15;

  const bf16*  wsW1 = (const bf16*)(ws + 0);
  const bf16*  wsW2 = (const bf16*)(ws + 8192);
  const bf16*  wsWc = (const bf16*)(ws + 16384);
  const bf16*  wsWm = (const bf16*)(ws + 24576);
  const float* wn   = (const float*)(ws + 57344);
  const float* bme  = (const float*)(ws + 58368);

  const float* capg = cap + (size_t)ci * 32 * 1024;
  const float* imgg = img + (size_t)ii * 36 * 1024;

  // ---- phase 0: caption -> LDS bf16; q-norms; zero logits accumulator -----
  for (int idx = tid; idx < 32 * 1024; idx += 256){
    int w = idx >> 10, d = idx & 1023;
    cap_sm[w * CAP_RS + d] = (bf16)capg[idx];
  }
  for (int idx = tid; idx < 48 * 64; idx += 256){
    int s = idx >> 6, nb = idx & 63;
    float acc = 0.f;
    if (s < 36){
      const float* rp = imgg + s * 1024 + nb * 16;
      #pragma unroll
      for (int e = 0; e < 16; ++e){ float v = rp[e]; acc += v * v; }
    }
    qn_sm[idx] = acc;
  }
  for (int idx = tid; idx < 32 * 48; idx += 256) a_sm[idx] = 0.f;
  __syncthreads();

  // ---- phase 1: a[w][s] = cap . img, M=32 N=48 K=1024 ---------------------
  // 24 balanced jobs: (wt:2) x (st:3) x (K-quarter:4); partials combined via
  // LDS float atomics (4 low-contention adds per element).
  for (int j = wv; j < 24; j += 8){
    int wt = j / 12, rem = j % 12, st = rem / 4, kq = rem % 4;
    int srow  = st * 16 + n;
    float msk = (srow < 36) ? 1.f : 0.f;                  // branch-free OOB row
    const float* brow = imgg + (srow < 36 ? srow : 35) * 1024;
    v8f acc = {0.f,0.f,0.f,0.f,0.f,0.f,0.f,0.f};
    for (int kc = kq * 8; kc < kq * 8 + 8; ++kc){
      v16bf a = ldsA(cap_sm, CAP_RS, wt * 16, kc * 32, lane);
      const float* bp = brow + kc * 32 + g * 16;
      v16bf b;
      #pragma unroll
      for (int e = 0; e < 16; ++e) b[e] = (bf16)(bp[e] * msk);
      acc = WMMA_BF16(a, b, acc);
    }
    #pragma unroll
    for (int r = 0; r < 8; ++r){
      int w = wt * 16 + r + 8 * g;
      atomicAdd(&a_sm[w * 48 + (st * 16 + n)], acc[r]);
    }
  }
  __syncthreads();
  // leaky-relu pass over accumulated logits
  for (int idx = tid; idx < 32 * 48; idx += 256){
    float v = a_sm[idx];
    a_sm[idx] = (v > 0.f) ? v : 0.1f * v;
  }
  __syncthreads();

  // ---- phase 2: l2norm over s, softmax over w -----------------------------
  if (tid < 32){
    float s2 = 0.f;
    for (int s = 0; s < 48; ++s){ float v = a_sm[tid * 48 + s]; s2 += v * v; }
    invn[tid] = 1.0f / (sqrtf(s2) + 1e-8f);
  }
  __syncthreads();
  if (tid < 48){
    int s = tid;
    float mx = -1e30f;
    float xv[32];
    #pragma unroll
    for (int w = 0; w < 32; ++w){
      float v = 9.0f * a_sm[w * 48 + s] * invn[w];
      xv[w] = v; mx = fmaxf(mx, v);
    }
    float se = 0.f;
    #pragma unroll
    for (int w = 0; w < 32; ++w){ xv[w] = __expf(xv[w] - mx); se += xv[w]; }
    float inv = 1.0f / se;
    #pragma unroll
    for (int w = 0; w < 32; ++w) attn_sm[s * 32 + w] = (bf16)(xv[w] * inv);
  }
  __syncthreads();

  // ---- phase 3: ctx tiles (attn @ cap) + blockwise cosine partials --------
  // st-major so each wave loads the attn A-fragment once per s-tile.
  for (int st = 0; st < 3; ++st){
    v16bf a = ldsA(attn_sm, 32, st * 16, 0, lane);
    for (int nb = wv; nb < 64; nb += 8){
      v16bf b;
      const bf16* pc = cap_sm + (size_t)(g * 16) * CAP_RS + nb * 16 + n;
      #pragma unroll
      for (int e = 0; e < 16; ++e) b[e] = pc[(size_t)e * CAP_RS];
      v8f acc = {0.f,0.f,0.f,0.f,0.f,0.f,0.f,0.f};
      acc = WMMA_BF16(a, b, acc);
      #pragma unroll
      for (int r = 0; r < 8; ++r){
        int s = st * 16 + r + 8 * g;
        float q  = (s < 36) ? imgg[s * 1024 + nb * 16 + n] : 0.f;
        float cb = acc[r];
        float p = q * cb, c2 = cb * cb;
        #pragma unroll
        for (int off = 1; off < 16; off <<= 1){
          p  += __shfl_xor(p,  off, 32);
          c2 += __shfl_xor(c2, off, 32);
        }
        if (n == 0){ w12_sm[s * 64 + nb] = p; ncb_sm[s * 64 + nb] = c2; }
      }
    }
  }
  __syncthreads();
  for (int idx = tid; idx < 48 * 64; idx += 256){
    int nb = idx & 63;
    float nrm = sqrtf(qn_sm[idx]) * sqrtf(ncb_sm[idx]);
    float sm = w12_sm[idx] / fmaxf(nrm, 1e-8f) * delt[nb];
    sim_sm[idx] = (bf16)sm;
  }
  __syncthreads();

  // ---- phase 4: f1 = sim@W1^T+b1 ; f2 = sim@W2^T+b2 (M=48 N=64 K=64) ------
  for (int j = wv; j < 24; j += 8){
    int which = j / 12, rem = j % 12;
    int st = rem / 4, nt = rem % 4;
    const bf16*  Wsrc = which ? wsW2 : wsW1;
    const float* bias = which ? b2 : b1;
    bf16* dst = which ? f2_sm : f1_sm;
    v8f acc = {0.f,0.f,0.f,0.f,0.f,0.f,0.f,0.f};
    #pragma unroll
    for (int ks = 0; ks < 2; ++ks){
      v16bf a = ldsA(sim_sm, 64, st * 16, ks * 32, lane);
      const bf16* row = Wsrc + (nt * 16 + n) * 64 + ks * 32 + g * 16;
      v16bf b;
      #pragma unroll
      for (int e = 0; e < 16; ++e) b[e] = row[e];
      acc = WMMA_BF16(a, b, acc);
    }
    int col = nt * 16 + n;
    float bv = bias[col];
    #pragma unroll
    for (int r = 0; r < 8; ++r){
      int s = st * 16 + r + 8 * g;
      dst[s * 64 + col] = (bf16)(acc[r] + bv);
    }
  }
  __syncthreads();

  // ---- phase 5: gram[s][t] = f1 . f2 (M=48 N=48 K=64) ---------------------
  for (int j = wv; j < 9; j += 8){
    int st = j / 3, nt = j % 3;
    v8f acc = {0.f,0.f,0.f,0.f,0.f,0.f,0.f,0.f};
    #pragma unroll
    for (int ks = 0; ks < 2; ++ks){
      v16bf a = ldsA(f1_sm, 64, st * 16, ks * 32, lane);
      const bf16* row = f2_sm + (nt * 16 + n) * 64 + ks * 32 + g * 16;
      v16bf b;
      #pragma unroll
      for (int e = 0; e < 16; ++e) b[e] = row[e];
      acc = WMMA_BF16(a, b, acc);
    }
    int t = nt * 16 + n;
    #pragma unroll
    for (int r = 0; r < 8; ++r){
      int s = st * 16 + r + 8 * g;
      gram_sm[s * 52 + t] = acc[r];
    }
  }
  __syncthreads();

  // ---- phase 6: softmax over t<36, zero-pad K to 64 -----------------------
  if (tid < 48){
    int s = tid;
    float mx = -1e30f;
    for (int t = 0; t < 36; ++t) mx = fmaxf(mx, gram_sm[s * 52 + t]);
    float se = 0.f;
    for (int t = 0; t < 36; ++t) se += __expf(gram_sm[s * 52 + t] - mx);
    float inv = 1.0f / se;
    for (int t = 0; t < 36; ++t) gw_sm[s * 64 + t] = (bf16)(__expf(gram_sm[s * 52 + t] - mx) * inv);
    for (int t = 36; t < 64; ++t) gw_sm[s * 64 + t] = (bf16)0.0f;
  }
  __syncthreads();

  // ---- phase 7: msg = gw @ sim (M=48 N=64 K=64-padded) --------------------
  for (int j = wv; j < 12; j += 8){
    int st = j / 4, nt = j % 4;
    v8f acc = {0.f,0.f,0.f,0.f,0.f,0.f,0.f,0.f};
    #pragma unroll
    for (int ks = 0; ks < 2; ++ks){
      v16bf a = ldsA(gw_sm, 64, st * 16, ks * 32, lane);
      v16bf b;
      #pragma unroll
      for (int e = 0; e < 16; ++e)
        b[e] = sim_sm[(ks * 32 + g * 16 + e) * 64 + nt * 16 + n];
      acc = WMMA_BF16(a, b, acc);
    }
    #pragma unroll
    for (int r = 0; r < 8; ++r){
      int s = st * 16 + r + 8 * g;
      msg_sm[s * 64 + nt * 16 + n] = (bf16)acc[r];
    }
  }
  __syncthreads();

  // ---- phase 8: conv = msg @ Wc + conv_b ----------------------------------
  for (int j = wv; j < 12; j += 8){
    int st = j / 4, nt = j % 4;
    v8f acc = {0.f,0.f,0.f,0.f,0.f,0.f,0.f,0.f};
    #pragma unroll
    for (int ks = 0; ks < 2; ++ks){
      v16bf a = ldsA(msg_sm, 64, st * 16, ks * 32, lane);
      const bf16* row = wsWc + (nt * 16 + n) * 64 + ks * 32 + g * 16;
      v16bf b;
      #pragma unroll
      for (int e = 0; e < 16; ++e) b[e] = row[e];
      acc = WMMA_BF16(a, b, acc);
    }
    int col = nt * 16 + n;
    float bv = convb[col];
    #pragma unroll
    for (int r = 0; r < 8; ++r){
      int s = st * 16 + r + 8 * g;
      conv_sm[s * 64 + col] = (bf16)(acc[r] + bv);
    }
  }
  __syncthreads();

  if (tid < 48) ysum[tid] = 0.f;
  __syncthreads();

  // ---- phase 9: h = conv @ WmT + bme; fused dot with wn -------------------
  for (int j = wv; j < 48; j += 8){
    int st = j / 16, ht = j % 16;
    v8f acc = {0.f,0.f,0.f,0.f,0.f,0.f,0.f,0.f};
    #pragma unroll
    for (int ks = 0; ks < 2; ++ks){
      v16bf a = ldsA(conv_sm, 64, st * 16, ks * 32, lane);
      const bf16* row = wsWm + (ht * 16 + n) * 64 + ks * 32 + g * 16;
      v16bf b;
      #pragma unroll
      for (int e = 0; e < 16; ++e) b[e] = row[e];
      acc = WMMA_BF16(a, b, acc);
    }
    int h = ht * 16 + n;
    float bmv = bme[h], wnv = wn[h];
    #pragma unroll
    for (int r = 0; r < 8; ++r){
      float contrib = (acc[r] + bmv) * wnv;
      #pragma unroll
      for (int off = 1; off < 16; off <<= 1)
        contrib += __shfl_xor(contrib, off, 32);
      if (n == 0) atomicAdd(&ysum[st * 16 + r + 8 * g], contrib);
    }
  }
  __syncthreads();

  // ---- phase 10: tanh, mean over regions ----------------------------------
  if (tid < 36) ytan[tid] = tanhf(ysum[tid] + bo[0]);
  __syncthreads();
  if (tid == 0){
    float acc = 0.f;
    for (int s = 0; s < 36; ++s) acc += ytan[s];
    out[ii * 48 + ci] = acc * (1.0f / 36.0f);
  }
}

extern "C" void kernel_launch(void* const* d_in, const int* in_sizes, int n_in,
                              void* d_out, int out_size, void* d_ws, size_t ws_size,
                              hipStream_t stream){
  (void)in_sizes; (void)n_in; (void)out_size; (void)ws_size;
  const float* img  = (const float*)d_in[0];
  const float* cap  = (const float*)d_in[1];
  // d_in[2] = cap_lens : unused (reference processes full length)
  const float* delt = (const float*)d_in[3];
  const float* W1   = (const float*)d_in[4];
  const float* b1   = (const float*)d_in[5];
  const float* W2   = (const float*)d_in[6];
  const float* b2   = (const float*)d_in[7];
  const float* cw   = (const float*)d_in[8];
  const float* cb   = (const float*)d_in[9];
  const float* Wm   = (const float*)d_in[10];
  const float* bm   = (const float*)d_in[11];
  const float* V    = (const float*)d_in[12];
  const float* gg   = (const float*)d_in[13];
  const float* bo   = (const float*)d_in[14];
  char*  ws  = (char*)d_ws;
  float* out = (float*)d_out;

  i2t_prep<<<1, 256, 0, stream>>>(W1, W2, cw, Wm, bm, V, gg, ws);
  i2t_main<<<dim3(48, 48), 256, 0, stream>>>(img, cap, delt, b1, b2, cb, bo, ws, out);
}
